// QuestionEncoderLSTM_82841329205716
// MI455X (gfx1250) — matrix operand order repaired and checked by reading, not compiled, not run on hardware
//
#include <hip/hip_runtime.h>

typedef _Float16 half_t;
typedef __attribute__((ext_vector_type(16))) _Float16 v16h;
typedef __attribute__((ext_vector_type(8)))  _Float16 v8h;
typedef __attribute__((ext_vector_type(8)))  float    v8f;

constexpr int Bn   = 64;     // batch
constexpr int Tn   = 32;     // sequence length
constexpr int En   = 300;    // embed dim
constexpr int EPAD = 320;    // embed dim padded to multiple of 32
constexpr int Hn   = 1024;   // hidden
constexpr int G4   = 4096;   // 4*H

// ---------------------------------------------------------------------------
// fp32 -> fp16 weight conversion with zero-padding of K to dstK
// ---------------------------------------------------------------------------
__global__ void convert_w(const float* __restrict__ src, half_t* __restrict__ dst,
                          int rows, int srcK, int dstK) {
    int idx = blockIdx.x * blockDim.x + threadIdx.x;
    if (idx >= rows * dstK) return;
    int r = idx / dstK, k = idx - r * dstK;
    float v = (k < srcK) ? src[r * srcK + k] : 0.0f;
    dst[idx] = (half_t)v;
}

// ---------------------------------------------------------------------------
// embedding gather -> time-major fp16 [T, B, EPAD] (zero padded cols)
// ---------------------------------------------------------------------------
__global__ void embed_k(const int* __restrict__ q, const float* __restrict__ table,
                        half_t* __restrict__ xh) {
    int idx = blockIdx.x * blockDim.x + threadIdx.x;   // T*B*EPAD exactly
    int k  = idx % EPAD;
    int bt = idx / EPAD;
    int b  = bt % Bn;
    int t  = bt / Bn;
    int tok = q[b * Tn + t];
    float v = (k < En) ? table[tok * En + k] : 0.0f;
    xh[idx] = (half_t)v;
}

// ---------------------------------------------------------------------------
// zero-init recurrent state (fp32 h/c and fp16 shadow of h)
// ---------------------------------------------------------------------------
__global__ void init_state(float* h1, float* c1, float* h2, float* c2,
                           half_t* h1h, half_t* h2h) {
    int i = blockIdx.x * blockDim.x + threadIdx.x;     // B*H exactly
    h1[i] = 0.f; c1[i] = 0.f; h2[i] = 0.f; c2[i] = 0.f;
    h1h[i] = (half_t)0.f; h2h[i] = (half_t)0.f;
}

// ---------------------------------------------------------------------------
// Dual-tile WMMA accumulation: one shared A fragment feeds two independent
// accumulator chains (two adjacent N-tiles), hiding the F16 WMMA->WMMA RAW
// hazard (~5 slots per ISA 7.12.1) inside the wave and halving A traffic.
// Fragment packing per CDNA5 ISA 16-bit layouts (wave32):
//   A: lane l holds row (l&15); K-halves chosen by kh=(l>>4):
//      a[0..7]  = A[row][kb + kh*8 + 0..7]
//      a[8..15] = A[row][kb + 16 + kh*8 + 0..7]
//   B: lane l holds col n=(l&15) (i.e. weight row n):
//      b[0..7]  = W[n][kb + kh*8 + 0..7]
//      b[8..15] = W[n][kb + 16 + kh*8 + 0..7]
// ---------------------------------------------------------------------------
__device__ __forceinline__ void mma_accum2(v8f& acc0, v8f& acc1,
                                           const half_t* __restrict__ aRow,
                                           const half_t* __restrict__ w0Row,
                                           const half_t* __restrict__ w1Row,
                                           int K, int kh) {
    for (int kb = 0; kb < K; kb += 32) {
        v8h alo = *(const v8h*)(aRow  + kb + kh * 8);
        v8h ahi = *(const v8h*)(aRow  + kb + 16 + kh * 8);
        v8h b0l = *(const v8h*)(w0Row + kb + kh * 8);
        v8h b0h = *(const v8h*)(w0Row + kb + 16 + kh * 8);
        v8h b1l = *(const v8h*)(w1Row + kb + kh * 8);
        v8h b1h = *(const v8h*)(w1Row + kb + 16 + kh * 8);
        v16h af, bf0, bf1;
#pragma unroll
        for (int i = 0; i < 8; ++i) {
            af[i]  = alo[i]; af[8 + i]  = ahi[i];
            bf0[i] = b0l[i]; bf0[8 + i] = b0h[i];
            bf1[i] = b1l[i]; bf1[8 + i] = b1h[i];
        }
        acc0 = __builtin_amdgcn_wmma_f32_16x16x32_f16(
            false, af, false, bf0, (short)0, acc0, false, false);
        acc1 = __builtin_amdgcn_wmma_f32_16x16x32_f16(
            false, af, false, bf1, (short)0, acc1, false, false);
    }
}

// ---------------------------------------------------------------------------
// gates = A0[64,K0] @ W0[4096,K0]^T + A1[64,K1] @ W1[4096,K1]^T
// grid = 64 blocks x 256 threads = 512 waves = (4 M-tiles) x (128 N-pairs);
// each wave owns a 16x32 output slab (two 16x16 WMMA tiles).
// ---------------------------------------------------------------------------
__launch_bounds__(256)
__global__ void gates_gemm(const half_t* __restrict__ A0, int K0, const half_t* __restrict__ W0,
                           const half_t* __restrict__ A1, int K1, const half_t* __restrict__ W1,
                           float* __restrict__ gates) {
    const int lane  = threadIdx.x & 31;
    const int wave  = blockIdx.x * (blockDim.x >> 5) + (threadIdx.x >> 5); // 0..511
    const int mtile = wave >> 7;    // 0..3
    const int npair = wave & 127;   // 0..127  -> N-tiles 2*npair, 2*npair+1
    const int r15   = lane & 15;
    const int kh    = lane >> 4;

    const int arow = mtile * 16 + r15;   // batch row
    const int n0   = npair * 32 + r15;   // weight row for N-tile 0
    const int n1   = n0 + 16;            // weight row for N-tile 1

    v8f acc0 = {}, acc1 = {};
    mma_accum2(acc0, acc1, A0 + (size_t)arow * K0,
               W0 + (size_t)n0 * K0, W0 + (size_t)n1 * K0, K0, kh);
    mma_accum2(acc0, acc1, A1 + (size_t)arow * K1,
               W1 + (size_t)n0 * K1, W1 + (size_t)n1 * K1, K1, kh);

    // C layout: VGPR r -> row (mtile*16 + 8*kh + r), col base npair*32 + r15
    float* g = gates + (size_t)(mtile * 16 + 8 * kh) * G4 + npair * 32 + r15;
#pragma unroll
    for (int r = 0; r < 8; ++r) {
        g[(size_t)r * G4]      = acc0[r];
        g[(size_t)r * G4 + 16] = acc1[r];
    }
}

// ---------------------------------------------------------------------------
// pointwise LSTM cell with length masking; updates fp32 h/c + fp16 shadow h
// ---------------------------------------------------------------------------
__global__ void lstm_cell_k(const float* __restrict__ gates,
                            const float* __restrict__ b_ih, const float* __restrict__ b_hh,
                            float* __restrict__ h, float* __restrict__ c,
                            half_t* __restrict__ hh,
                            const int* __restrict__ lens, int t) {
    int idx = blockIdx.x * blockDim.x + threadIdx.x;   // B*H exactly
    int b = idx >> 10;
    int j = idx & 1023;
    if (t >= lens[b]) return;                          // frozen state past length
    const float* g = gates + (size_t)b * G4;
    float gi = g[j]        + b_ih[j]        + b_hh[j];
    float gf = g[1024 + j] + b_ih[1024 + j] + b_hh[1024 + j];
    float gg = g[2048 + j] + b_ih[2048 + j] + b_hh[2048 + j];
    float go = g[3072 + j] + b_ih[3072 + j] + b_hh[3072 + j];
    float si = 1.f / (1.f + expf(-gi));
    float sf = 1.f / (1.f + expf(-gf));
    float so = 1.f / (1.f + expf(-go));
    float cn = sf * c[idx] + si * tanhf(gg);
    float hn = so * tanhf(cn);
    c[idx] = cn;
    h[idx] = hn;
    hh[idx] = (half_t)hn;
}

// ---------------------------------------------------------------------------
// outputs: E_q [B,H,28,28] (h2 broadcast) ++ h1 ++ h2 ++ c1 ++ c2 (each B*H)
// ---------------------------------------------------------------------------
__global__ void write_out(const float* __restrict__ h1, const float* __restrict__ h2,
                          const float* __restrict__ c1, const float* __restrict__ c2,
                          float* __restrict__ out) {
    int idx = blockIdx.x * blockDim.x + threadIdx.x;   // total exactly
    const int total1 = Bn * Hn * 784;
    if (idx < total1) {
        int b = idx / (Hn * 784);
        int j = (idx / 784) % Hn;
        out[idx] = h2[b * Hn + j];
    } else {
        int k   = idx - total1;
        int sel = k / (Bn * Hn);
        int r   = k % (Bn * Hn);
        const float* src = (sel == 0) ? h1 : (sel == 1) ? h2 : (sel == 2) ? c1 : c2;
        out[idx] = src[r];
    }
}

// ---------------------------------------------------------------------------
extern "C" void kernel_launch(void* const* d_in, const int* in_sizes, int n_in,
                              void* d_out, int out_size, void* d_ws, size_t ws_size,
                              hipStream_t stream) {
    const int*   q     = (const int*)d_in[0];
    const int*   lens  = (const int*)d_in[1];
    const float* table = (const float*)d_in[2];
    const float* Wih1  = (const float*)d_in[3];
    const float* Whh1  = (const float*)d_in[4];
    const float* bih1  = (const float*)d_in[5];
    const float* bhh1  = (const float*)d_in[6];
    const float* Wih2  = (const float*)d_in[7];
    const float* Whh2  = (const float*)d_in[8];
    const float* bih2  = (const float*)d_in[9];
    const float* bhh2  = (const float*)d_in[10];

    // ---- workspace layout (256B aligned chunks, ~31 MB total) ----
    char* wp = (char*)d_ws;
    auto carve = [&](size_t bytes) -> char* {
        char* p = wp;
        wp += (bytes + 255) & ~(size_t)255;
        return p;
    };
    half_t* Wih1h  = (half_t*)carve((size_t)G4 * EPAD * sizeof(half_t));
    half_t* Whh1h  = (half_t*)carve((size_t)G4 * Hn   * sizeof(half_t));
    half_t* Wih2h  = (half_t*)carve((size_t)G4 * Hn   * sizeof(half_t));
    half_t* Whh2h  = (half_t*)carve((size_t)G4 * Hn   * sizeof(half_t));
    half_t* xh     = (half_t*)carve((size_t)Tn * Bn * EPAD * sizeof(half_t));
    float*  gates1 = (float*)carve((size_t)Bn * G4 * sizeof(float));
    float*  gates2 = (float*)carve((size_t)Bn * G4 * sizeof(float));
    float*  h1     = (float*)carve((size_t)Bn * Hn * sizeof(float));
    float*  c1     = (float*)carve((size_t)Bn * Hn * sizeof(float));
    float*  h2     = (float*)carve((size_t)Bn * Hn * sizeof(float));
    float*  c2     = (float*)carve((size_t)Bn * Hn * sizeof(float));
    half_t* h1h    = (half_t*)carve((size_t)Bn * Hn * sizeof(half_t));
    half_t* h2h    = (half_t*)carve((size_t)Bn * Hn * sizeof(half_t));

    // ---- one-time (per launch) preprocessing ----
    convert_w<<<(G4 * EPAD + 255) / 256, 256, 0, stream>>>(Wih1, Wih1h, G4, En, EPAD);
    convert_w<<<(G4 * Hn + 255) / 256, 256, 0, stream>>>(Whh1, Whh1h, G4, Hn, Hn);
    convert_w<<<(G4 * Hn + 255) / 256, 256, 0, stream>>>(Wih2, Wih2h, G4, Hn, Hn);
    convert_w<<<(G4 * Hn + 255) / 256, 256, 0, stream>>>(Whh2, Whh2h, G4, Hn, Hn);
    embed_k<<<(Tn * Bn * EPAD) / 256, 256, 0, stream>>>(q, table, xh);
    init_state<<<(Bn * Hn) / 256, 256, 0, stream>>>(h1, c1, h2, c2, h1h, h2h);

    // ---- recurrent chain: 32 steps x (gemm, cell) x 2 layers ----
    for (int t = 0; t < Tn; ++t) {
        gates_gemm<<<64, 256, 0, stream>>>(xh + (size_t)t * Bn * EPAD, EPAD, Wih1h,
                                           h1h, Hn, Whh1h, gates1);
        lstm_cell_k<<<(Bn * Hn) / 256, 256, 0, stream>>>(gates1, bih1, bhh1,
                                                         h1, c1, h1h, lens, t);
        gates_gemm<<<64, 256, 0, stream>>>(h1h, Hn, Wih2h,
                                           h2h, Hn, Whh2h, gates2);
        lstm_cell_k<<<(Bn * Hn) / 256, 256, 0, stream>>>(gates2, bih2, bhh2,
                                                         h2, c2, h2h, lens, t);
    }

    // ---- final broadcast + state concat (exact multiple of 256) ----
    const int total = Bn * Hn * 784 + 4 * Bn * Hn;     // 51,642,368
    write_out<<<total / 256, 256, 0, stream>>>(h1, h2, c1, c2, (float*)d_out);
}
